// DFSMNLayer_34127810134561
// MI455X (gfx1250) — matrix operand order repaired and checked by reading, not compiled, not run on hardware
//
#include <hip/hip_runtime.h>
#include <hip/hip_bf16.h>

typedef __bf16 bf16;
typedef bf16  v16bf __attribute__((ext_vector_type(16)));
typedef float v8f   __attribute__((ext_vector_type(8)));

union Frag {
    v16bf v;
    uint4 q[2];
};

#define BLK_M 128
#define BLK_N 64
#define BLK_K 32
#define LDSW 40   // bf16 elems per LDS row: 32 data + 8 pad => 80B row stride

// ---------------------------------------------------------------------------
// CDNA5 async global->LDS copy (16B), tracked by ASYNCcnt.
// saddr form: vdst = LDS byte address, vaddr = 32-bit byte offset, saddr = base.
// ---------------------------------------------------------------------------
__device__ __forceinline__
void async_copy_b128(uint32_t lds_byte, const bf16* base, uint32_t goff_byte)
{
    asm volatile("global_load_async_to_lds_b128 %0, %1, %2"
                 :: "v"(lds_byte), "v"(goff_byte),
                    "s"((unsigned long long)(uintptr_t)base)
                 : "memory");
}

// ---------------------------------------------------------------------------
// Split fp32 -> bf16 hi/lo pair (bf16x3 GEMM emulation inputs)
// ---------------------------------------------------------------------------
__global__ __launch_bounds__(256)
void split_fp32_bf16(const float* __restrict__ src, bf16* __restrict__ hi,
                     bf16* __restrict__ lo, int n)
{
    int base = blockIdx.x * 1024 + threadIdx.x;
    #pragma unroll
    for (int k = 0; k < 4; ++k) {
        int idx = base + k * 256;
        if (idx < n) {
            float x = src[idx];
            bf16 h = (bf16)x;
            hi[idx] = h;
            lo[idx] = (bf16)(x - (float)h);
        }
    }
}

// ---------------------------------------------------------------------------
// taps[j] = sum_h mem_w[j][h] (j<50), taps[50+j] = sum_h la_w[j][h]
// ---------------------------------------------------------------------------
__global__ __launch_bounds__(256)
void tap_sums(const float* __restrict__ mem_w, const float* __restrict__ la_w,
              float* __restrict__ taps, int Hd)
{
    int j = blockIdx.x;  // 0..54
    const float* src = (j < 50) ? (mem_w + (size_t)j * Hd)
                                : (la_w + (size_t)(j - 50) * Hd);
    float s = 0.0f;
    for (int c = threadIdx.x; c < Hd; c += 256) s += src[c];
    #pragma unroll
    for (int off = 16; off >= 1; off >>= 1) s += __shfl_xor(s, off, 32);
    __shared__ float red[8];
    if ((threadIdx.x & 31) == 0) red[threadIdx.x >> 5] = s;
    __syncthreads();
    if (threadIdx.x == 0) {
        float tot = 0.0f;
        #pragma unroll
        for (int w = 0; w < 8; ++w) tot += red[w];
        taps[j] = tot;
    }
}

// ---------------------------------------------------------------------------
// H_out[m][n] = sum_k A[m][k]*B[n][k] + bias[n]   (NT GEMM, bf16x3 via WMMA)
// Double-buffered LDS, async global->LDS staging overlapped with WMMA.
// ---------------------------------------------------------------------------
__global__ __launch_bounds__(256)
void gemm_bf16x3(const bf16* __restrict__ Ahi, const bf16* __restrict__ Alo,
                 const bf16* __restrict__ Bhi, const bf16* __restrict__ Blo,
                 const float* __restrict__ bias, float* __restrict__ H_out,
                 int M, int N, int K)
{
    __shared__ bf16 lA[2][2][BLK_M * LDSW];   // [buf][hi/lo][row*LDSW + k]
    __shared__ bf16 lB[2][2][BLK_N * LDSW];

    const int tid  = threadIdx.x;
    const int lane = tid & 31;
    const int wid  = tid >> 5;
    const int wm   = wid & 3;     // wave M index: 0..3
    const int wn   = wid >> 2;    // wave N index: 0..1

    const int gm0 = blockIdx.y * BLK_M;
    const int gn0 = blockIdx.x * BLK_N;

    v8f acc[2][2] = {};

    const int arow  = lane & 15;
    const int akoff = (lane < 16) ? 0 : 8;    // A: two 8-elem chunks per lane
    const int bkoff = (lane < 16) ? 0 : 8;    // B: two 8-elem chunks per lane

    // Issue one K-stage of async copies: 6 x b128 per thread (in-order done).
    auto issue = [&](int kb, int buf) {
        uint32_t la0 = (uint32_t)(uintptr_t)(&lA[buf][0][0]);
        uint32_t la1 = (uint32_t)(uintptr_t)(&lA[buf][1][0]);
        uint32_t lb0 = (uint32_t)(uintptr_t)(&lB[buf][0][0]);
        uint32_t lb1 = (uint32_t)(uintptr_t)(&lB[buf][1][0]);
        #pragma unroll
        for (int p = 0; p < 2; ++p) {                       // A: 512 chunks/mat
            int u = tid + p * 256;
            int r = u >> 2, c = (u & 3) * 8;
            uint32_t go  = (uint32_t)(((gm0 + r) * K + kb + c) * (int)sizeof(bf16));
            uint32_t off = (uint32_t)((r * LDSW + c) * (int)sizeof(bf16));
            async_copy_b128(la0 + off, Ahi, go);
            async_copy_b128(la1 + off, Alo, go);
        }
        {                                                    // B: 256 chunks/mat
            int r = tid >> 2, c = (tid & 3) * 8;
            uint32_t go  = (uint32_t)(((gn0 + r) * K + kb + c) * (int)sizeof(bf16));
            uint32_t off = (uint32_t)((r * LDSW + c) * (int)sizeof(bf16));
            async_copy_b128(lb0 + off, Bhi, go);
            async_copy_b128(lb1 + off, Blo, go);
        }
    };

    const int nk = K / BLK_K;
    issue(0, 0);                               // prologue: stage 0 -> buf 0

    for (int i = 0; i < nk; ++i) {
        const int buf = i & 1;
        __syncthreads();                       // buf^1 readers (iter i-1) done
        if (i + 1 < nk) {
            issue((i + 1) * BLK_K, buf ^ 1);   // overlap with this stage's math
            asm volatile("s_wait_asynccnt 0x6" ::: "memory");  // stage i landed
        } else {
            asm volatile("s_wait_asynccnt 0x0" ::: "memory");
        }
        __syncthreads();                       // publish stage i to all waves

        const bf16* LAh = lA[buf][0];
        const bf16* LAl = lA[buf][1];
        const bf16* LBh = lB[buf][0];
        const bf16* LBl = lB[buf][1];

        Frag ahi[2], alo[2], bhi[2], blo[2];
        #pragma unroll
        for (int mi = 0; mi < 2; ++mi) {
            int r = wm * 32 + mi * 16 + arow;
            ahi[mi].q[0] = *(const uint4*)(&LAh[r * LDSW + akoff]);
            ahi[mi].q[1] = *(const uint4*)(&LAh[r * LDSW + akoff + 16]);
            alo[mi].q[0] = *(const uint4*)(&LAl[r * LDSW + akoff]);
            alo[mi].q[1] = *(const uint4*)(&LAl[r * LDSW + akoff + 16]);
        }
        #pragma unroll
        for (int ni = 0; ni < 2; ++ni) {
            int r = wn * 32 + ni * 16 + (lane & 15);
            bhi[ni].q[0] = *(const uint4*)(&LBh[r * LDSW + bkoff]);
            bhi[ni].q[1] = *(const uint4*)(&LBh[r * LDSW + bkoff + 16]);
            blo[ni].q[0] = *(const uint4*)(&LBl[r * LDSW + bkoff]);
            blo[ni].q[1] = *(const uint4*)(&LBl[r * LDSW + bkoff + 16]);
        }

        #pragma unroll
        for (int mi = 0; mi < 2; ++mi) {
            #pragma unroll
            for (int ni = 0; ni < 2; ++ni) {
                acc[mi][ni] = __builtin_amdgcn_wmma_f32_16x16x32_bf16(
                    false, ahi[mi].v, false, bhi[ni].v, (short)0, acc[mi][ni], false, false);
                acc[mi][ni] = __builtin_amdgcn_wmma_f32_16x16x32_bf16(
                    false, ahi[mi].v, false, blo[ni].v, (short)0, acc[mi][ni], false, false);
                acc[mi][ni] = __builtin_amdgcn_wmma_f32_16x16x32_bf16(
                    false, alo[mi].v, false, bhi[ni].v, (short)0, acc[mi][ni], false, false);
            }
        }
    }

    // epilogue: C layout (lane<16 -> M=v, lane>=16 -> M=v+8), add bias
    const int mhalf = (lane < 16) ? 0 : 8;
    const int ccol  = lane & 15;
    #pragma unroll
    for (int mi = 0; mi < 2; ++mi) {
        #pragma unroll
        for (int ni = 0; ni < 2; ++ni) {
            int col  = gn0 + wn * 32 + ni * 16 + ccol;
            int rowb = gm0 + wm * 32 + mi * 16 + mhalf;
            float bv = bias[col];
            #pragma unroll
            for (int v = 0; v < 8; ++v) {
                H_out[(size_t)(rowb + v) * N + col] = acc[mi][ni][v] + bv;
            }
        }
    }
}

// ---------------------------------------------------------------------------
// y[b,t,c] = h[t] + mem[t] + fut[t], temporally tiled (T=64, C=128) in LDS.
//   t >= 50 : mem[t] = sum_{j<50} wm[j] * h[t-50+j]
//   t <  50 : mem[t] = sum_{j<t}  wm[j] * h[j]    (only in each batch's 1st tile)
//   fut[t]  = sum_{j<5, t+1+j<L} wf[j] * h[t+1+j]
// ---------------------------------------------------------------------------
#define CT 64             // timesteps per block (divides L)
#define CC 128            // channels per block
#define CW (CT + 55)      // staged window rows

__global__ __launch_bounds__(256)
void ctx_fuse(const float* __restrict__ h, const float* __restrict__ taps,
              float* __restrict__ y, int L, int Hd)
{
    __shared__ float sh[CW][CC];     // ~61 KB
    __shared__ float s_taps[64];

    const int tid  = threadIdx.x;
    const int t0g  = blockIdx.x * CT;          // global row start
    const int c0   = blockIdx.y * CC;
    const int b    = t0g / L;
    const int t0   = t0g - b * L;              // within-batch start
    const int wbase = t0 - 50;

    if (tid < 55) s_taps[tid] = taps[tid];

    int lo = t0 - 50; if (lo < 0) lo = 0;
    int hi = t0 + CT + 5; if (hi > L) hi = L;
    const int nrows = hi - lo;
    const float* hb = h + ((size_t)b * L) * Hd + c0;
    for (int idx = tid; idx < nrows * CC; idx += 256) {
        int rr = idx >> 7;                     // CC == 128
        int cc = idx & (CC - 1);
        sh[lo + rr - wbase][cc] = hb[(size_t)(lo + rr) * Hd + cc];
    }
    __syncthreads();

    const int cc = tid & (CC - 1);
    float* yb = y + ((size_t)b * L) * Hd + c0;
    for (int ti = tid >> 7; ti < CT; ti += 2) {   // t uniform per wave
        const int t = t0 + ti;
        const int wrow = t - wbase;               // = ti + 50
        float acc = sh[wrow][cc];
        if (t >= 50) {
            #pragma unroll 5
            for (int j = 0; j < 50; ++j)
                acc += s_taps[j] * sh[wrow - 50 + j][cc];
        } else {
            for (int j = 0; j < t; ++j)
                acc += s_taps[j] * sh[j - wbase][cc];
        }
        int nf = L - 1 - t; if (nf > 5) nf = 5;
        for (int j = 0; j < nf; ++j)
            acc += s_taps[50 + j] * sh[wrow + 1 + j][cc];
        yb[(size_t)t * Hd + cc] = acc;
    }
}

// ---------------------------------------------------------------------------
// In-place LayerNorm over the last dim (one block per row).
// ---------------------------------------------------------------------------
__global__ __launch_bounds__(256)
void layernorm_inplace(float* __restrict__ io, const float* __restrict__ gamma,
                       const float* __restrict__ beta, int Hd)
{
    __shared__ float s_red[16];
    const int tid = threadIdx.x;
    float* p = io + (size_t)blockIdx.x * Hd;

    float v[8];
    float s = 0.0f, ss = 0.0f;
    #pragma unroll
    for (int i = 0; i < 8; ++i) {
        v[i] = p[tid + i * 256];
        s += v[i]; ss += v[i] * v[i];
    }
    #pragma unroll
    for (int off = 16; off >= 1; off >>= 1) {
        s  += __shfl_xor(s,  off, 32);
        ss += __shfl_xor(ss, off, 32);
    }
    if ((tid & 31) == 0) { s_red[tid >> 5] = s; s_red[8 + (tid >> 5)] = ss; }
    __syncthreads();
    float S = 0.0f, SS = 0.0f;
    #pragma unroll
    for (int w = 0; w < 8; ++w) { S += s_red[w]; SS += s_red[8 + w]; }
    const float rH  = 1.0f / (float)Hd;
    float mu  = S * rH;
    float inv = rsqrtf(SS * rH - mu * mu + 1e-5f);
    #pragma unroll
    for (int i = 0; i < 8; ++i) {
        int c = tid + i * 256;
        p[c] = (v[i] - mu) * inv * gamma[c] + beta[c];
    }
}

// ---------------------------------------------------------------------------
extern "C" void kernel_launch(void* const* d_in, const int* in_sizes, int n_in,
                              void* d_out, int out_size, void* d_ws, size_t ws_size,
                              hipStream_t stream)
{
    const float* x     = (const float*)d_in[0];
    const float* W     = (const float*)d_in[1];
    const float* b_lin = (const float*)d_in[2];
    const float* mem_w = (const float*)d_in[3];
    const float* la_w  = (const float*)d_in[4];
    const float* gamma = (const float*)d_in[5];
    const float* beta  = (const float*)d_in[6];
    float* out = (float*)d_out;

    const int B = 8, L = 2048, D = 1024, Hd = 2048;
    const int M = B * L;          // 16384

    char* ws = (char*)d_ws;
    size_t off = 0;
    float* h   = (float*)(ws + off); off += (size_t)M * Hd * sizeof(float);
    bf16* xhi  = (bf16*)(ws + off);  off += (size_t)M * D * sizeof(bf16);
    bf16* xlo  = (bf16*)(ws + off);  off += (size_t)M * D * sizeof(bf16);
    bf16* whi  = (bf16*)(ws + off);  off += (size_t)Hd * D * sizeof(bf16);
    bf16* wlo  = (bf16*)(ws + off);  off += (size_t)Hd * D * sizeof(bf16);
    float* taps = (float*)(ws + off);

    int nx = M * D;      // 16,777,216
    int nw = Hd * D;     //  2,097,152
    split_fp32_bf16<<<(nx + 1023) / 1024, 256, 0, stream>>>(x, xhi, xlo, nx);
    split_fp32_bf16<<<(nw + 1023) / 1024, 256, 0, stream>>>(W, whi, wlo, nw);
    tap_sums<<<55, 256, 0, stream>>>(mem_w, la_w, taps, Hd);

    dim3 ggrid(Hd / BLK_N, M / BLK_M);   // (32, 128)
    gemm_bf16x3<<<ggrid, 256, 0, stream>>>(xhi, xlo, whi, wlo, b_lin, h, M, Hd, D);

    dim3 cgrid(M / CT, Hd / CC);         // (256, 16)
    ctx_fuse<<<cgrid, 256, 0, stream>>>(h, taps, out, L, Hd);

    layernorm_inplace<<<M, 256, 0, stream>>>(out, gamma, beta, Hd);
}
